// dLSTM_22978075033757
// MI455X (gfx1250) — compile-verified, matrix operand order checked by reading, not run on hardware
//
#include <hip/hip_runtime.h>
#include <hip/hip_bf16.h>

// ---------------------------------------------------------------------------
// Persistent 2-layer LSTM + per-step Bayesian linear for MI455X (gfx1250).
// bf16 WMMA (v_wmma_f32_16x16x32_bf16) with f32 accumulate; weights staged
// once into LDS per workgroup; cell state kept in registers; 3 grid barriers
// per timestep; activations ping-pong in bf16 workspace buffers.
// K-loops use explicit two-stage ping-pong fragment buffers (no rotation
// copies -> no WMMA hazard NOPs); global_prefetch of next step's eps_w slice;
// non-temporal eps loads.
// ---------------------------------------------------------------------------

constexpr int kH   = 512;   // hidden
constexpr int kB   = 64;    // batch
constexpr int kT   = 256;   // seq len
constexpr int kNWG = 32;    // persistent workgroups (1 per WGP, <= WGP count)
constexpr int kTPB = 512;   // 16 waves of 32
constexpr int kK2  = 1024;  // concat K = [x | h]
constexpr int kKP  = 1032;  // padded LDS row stride (bf16) for weights
constexpr int kKP2 = 520;   // padded LDS row stride (bf16) for bayes tile
constexpr int kXHN = kB * kK2; // ushorts per activation buffer

typedef __attribute__((ext_vector_type(16))) __bf16 v16bf;
typedef __attribute__((ext_vector_type(8)))  float  v8f;
typedef __attribute__((ext_vector_type(4)))  float  v4f;   // native vec for NT loads

union FragU { uint4 q[2]; v16bf bf; };

// LDS carve-up (bytes)
constexpr int SZ_W      = 4 * 16 * kKP * 2;            // 132096 per layer
constexpr int OFF_W0    = 0;
constexpr int OFF_W1    = OFF_W0 + SZ_W;               // 132096
constexpr int OFF_GBUF  = OFF_W1 + SZ_W;               // 264192
constexpr int OFF_WTILE = OFF_GBUF + 4 * kB * 16 * 4;  // 280576
constexpr int OFF_BC0   = OFF_WTILE + 16 * kKP2 * 2;   // 297216
constexpr int OFF_BC1   = OFF_BC0 + 64 * 4;            // 297472
constexpr int OFF_BB    = OFF_BC1 + 64 * 4;            // 297728
constexpr int kSMEM     = OFF_BB + 16 * 4;             // 297792 (< 320 KB WGP LDS)

__device__ inline unsigned short f2bf(float f) {
    unsigned u = __float_as_uint(f);
    u += 0x7fffu + ((u >> 16) & 1u);   // round-to-nearest-even
    return (unsigned short)(u >> 16);
}

__device__ inline v16bf ldfrag(const unsigned short* p) {
    FragU f;
    f.q[0] = *(const uint4*)(p);        // K group 0 (8 bf16)
    f.q[1] = *(const uint4*)(p + 16);   // K group 1 at k+16 (8 bf16)
    return f.bf;
}

__device__ inline float sigmoidf(float x) { return 1.0f / (1.0f + __expf(-x)); }
__device__ inline float tanh_fast(float x) {
    x = fminf(fmaxf(x, -15.0f), 15.0f);
    float e = __expf(2.0f * x);
    return (e - 1.0f) / (e + 1.0f);
}

// centralized monotonic-counter grid barrier (all kNWG WGs co-resident)
__device__ inline void grid_barrier(unsigned* bar, int& gen) {
    ++gen;
    __threadfence();              // release: make this WG's stores visible
    __syncthreads();
    if (threadIdx.x == 0) {
        atomicAdd(bar, 1u);
        const unsigned tgt = (unsigned)(gen * kNWG);
        while (__hip_atomic_load(bar, __ATOMIC_RELAXED, __HIP_MEMORY_SCOPE_AGENT) < tgt)
            __builtin_amdgcn_s_sleep(1);
    }
    __syncthreads();
    __threadfence();              // acquire: invalidate stale L0 lines in all waves
}

// one 16x16 gate tile per wave, K = 1024 ([x|h] concat), bf16 WMMA.
// Two-stage ping-pong: each WMMA consumes one fragment set while the loads
// for the other set are in flight -- no register rotation copies.
__device__ inline void gate_tiles(const unsigned short* xh, const unsigned short* WL,
                                  float* gbuf, int wv, int lane) {
    const int q  = wv >> 2;           // gate (i,f,g,o)
    const int mt = wv & 3;            // batch tile
    const int hi = lane >> 4;
    const unsigned short* A0 = xh + (16 * mt + (lane & 15)) * kK2 + hi * 8;
    const unsigned short* B0 = WL + (q * 16 + (lane & 15)) * kKP + hi * 8;
    v8f acc = {};
    v16bf aA = ldfrag(A0);
    v16bf bA = ldfrag(B0);
#pragma unroll 2
    for (int kk = 0; kk < kK2; kk += 64) {
        v16bf aB = ldfrag(A0 + kk + 32);
        v16bf bB = ldfrag(B0 + kk + 32);
        acc = __builtin_amdgcn_wmma_f32_16x16x32_bf16(false, aA, false, bA,
                                                      (short)0, acc, false, false);
        if (kk + 64 < kK2) {
            aA = ldfrag(A0 + kk + 64);
            bA = ldfrag(B0 + kk + 64);
        }
        acc = __builtin_amdgcn_wmma_f32_16x16x32_bf16(false, aB, false, bB,
                                                      (short)0, acc, false, false);
    }
    const int jj = lane & 15;
    const int mb = 16 * mt + hi * 8;
#pragma unroll
    for (int v = 0; v < 8; ++v)
        gbuf[q * 1024 + (mb + v) * 16 + jj] = acc[v];   // [q][b][jj]
}

__global__ void prep_kernel(const float* __restrict__ w_log_var,
                            const float* __restrict__ b_log_var,
                            float* __restrict__ sigw,
                            float* __restrict__ bsig,
                            unsigned* __restrict__ bar) {
    const int i = blockIdx.x * 256 + threadIdx.x;
    if (i == 0) *bar = 0u;
    if (i < kH) bsig[i] = __expf(0.5f * b_log_var[i]);
    if (i < kH * kH) sigw[i] = __expf(0.5f * w_log_var[i]);
}

__global__ __launch_bounds__(kTPB, 1)
void lstm_persistent(const float* __restrict__ init_h, const float* __restrict__ init_c,
                     const float* __restrict__ eps_w,  const float* __restrict__ eps_b,
                     const float* __restrict__ W_ih0,  const float* __restrict__ W_hh0,
                     const float* __restrict__ b_ih0,  const float* __restrict__ b_hh0,
                     const float* __restrict__ W_ih1,  const float* __restrict__ W_hh1,
                     const float* __restrict__ b_ih1,  const float* __restrict__ b_hh1,
                     const float* __restrict__ w_mu,   const float* __restrict__ b_mu,
                     const float* __restrict__ sigw,   const float* __restrict__ bsig,
                     unsigned* bar, unsigned short* xh0, unsigned short* xh1,
                     float* __restrict__ out) {
    extern __shared__ char smem[];
    unsigned short* WL0   = (unsigned short*)(smem + OFF_W0);
    unsigned short* WL1   = (unsigned short*)(smem + OFF_W1);
    float*          gbuf  = (float*)(smem + OFF_GBUF);
    unsigned short* wtile = (unsigned short*)(smem + OFF_WTILE);
    float*          bc0   = (float*)(smem + OFF_BC0);
    float*          bc1   = (float*)(smem + OFF_BC1);
    float*          bb    = (float*)(smem + OFF_BB);

    const int tid  = threadIdx.x;
    const int g    = blockIdx.x;      // owns hidden cols [16g, 16g+16)
    const int wv   = tid >> 5;
    const int lane = tid & 31;
    int gen = 0;

    // ---- stage bf16 weight slices [W_ih | W_hh] for both layers into LDS ----
    for (int l = 0; l < 2; ++l) {
        const float* Wih = l ? W_ih1 : W_ih0;
        const float* Whh = l ? W_hh1 : W_hh0;
        unsigned short* WL = l ? WL1 : WL0;
#pragma unroll 4
        for (int e = 0; e < 128; ++e) {
            const int flat = e * kTPB + tid;       // 0..65535
            const int q  = flat >> 14;
            const int jj = (flat >> 10) & 15;
            const int k  = flat & 1023;
            const int n  = q * kH + 16 * g + jj;   // gate row in [4H]
            const float v = (k < kH) ? Wih[n * kH + k] : Whh[n * kH + (k - kH)];
            WL[(q * 16 + jj) * kKP + k] = f2bf(v);
        }
    }
    if (tid < 64) {
        const int q = tid >> 4, jj = tid & 15;
        const int n = q * kH + 16 * g + jj;
        bc0[tid] = b_ih0[n] + b_hh0[n];
        bc1[tid] = b_ih1[n] + b_hh1[n];
    }

    // ---- init activations (buffer parity 0) and register-resident cell state ----
    float c0r[2], c1r[2];
#pragma unroll
    for (int p = 0; p < 2; ++p) {
        const int idx = tid + p * kTPB;
        const int b = idx >> 4, jj = idx & 15, col = 16 * g + jj;
        c0r[p] = init_c[0 * kB * kH + b * kH + col];
        c1r[p] = init_c[1 * kB * kH + b * kH + col];
        xh0[b * kK2 + col]      = 0;                                         // x0 = 0
        xh0[b * kK2 + kH + col] = f2bf(init_h[0 * kB * kH + b * kH + col]);
        xh1[b * kK2 + kH + col] = f2bf(init_h[1 * kB * kH + b * kH + col]);
    }
    grid_barrier(bar, gen);

    // ---- time loop: L0 -> cell0 -> | -> L1 -> cell1 -> | -> bayes -> | ----
    for (int s = 0; s < kT; ++s) {
        const unsigned short* xh0cur = xh0 + (s & 1) * kXHN;
        unsigned short*       xh1cur = xh1 + (s & 1) * kXHN;
        unsigned short*       xh0nxt = xh0 + ((s + 1) & 1) * kXHN;
        unsigned short*       xh1nxt = xh1 + ((s + 1) & 1) * kXHN;

        // prefetch next step's eps_w slice (32 KB per WG = 256 cachelines);
        // landing window = the whole L0+L1 compute of this step.
        if (s + 1 < kT && tid < 256) {
            const float* pf = eps_w + (size_t)(s + 1) * kH * kH
                                    + (size_t)(16 * g) * kH + tid * 32;
            __builtin_prefetch(pf, 0, 1);
        }

        // -------- layer 0 gates --------
        gate_tiles(xh0cur, WL0, gbuf, wv, lane);
        __syncthreads();
        // -------- cell 0 (c in registers) --------
#pragma unroll
        for (int p = 0; p < 2; ++p) {
            const int idx = tid + p * kTPB;
            const int b = idx >> 4, jj = idx & 15, col = 16 * g + jj;
            const float i_ = sigmoidf(gbuf[0 * 1024 + idx] + bc0[0 * 16 + jj]);
            const float f_ = sigmoidf(gbuf[1 * 1024 + idx] + bc0[1 * 16 + jj]);
            const float g_ = tanh_fast(gbuf[2 * 1024 + idx] + bc0[2 * 16 + jj]);
            const float o_ = sigmoidf(gbuf[3 * 1024 + idx] + bc0[3 * 16 + jj]);
            const float c  = f_ * c0r[p] + i_ * g_;
            c0r[p] = c;
            const unsigned short hb = f2bf(o_ * tanh_fast(c));
            xh1cur[b * kK2 + col]      = hb;   // layer-1 A input (x half)
            xh0nxt[b * kK2 + kH + col] = hb;   // next step layer-0 (h half)
        }
        grid_barrier(bar, gen);

        // -------- layer 1 gates --------
        gate_tiles(xh1cur, WL1, gbuf, wv, lane);
        __syncthreads();
        // -------- cell 1 --------
#pragma unroll
        for (int p = 0; p < 2; ++p) {
            const int idx = tid + p * kTPB;
            const int b = idx >> 4, jj = idx & 15, col = 16 * g + jj;
            const float i_ = sigmoidf(gbuf[0 * 1024 + idx] + bc1[0 * 16 + jj]);
            const float f_ = sigmoidf(gbuf[1 * 1024 + idx] + bc1[1 * 16 + jj]);
            const float g_ = tanh_fast(gbuf[2 * 1024 + idx] + bc1[2 * 16 + jj]);
            const float o_ = sigmoidf(gbuf[3 * 1024 + idx] + bc1[3 * 16 + jj]);
            const float c  = f_ * c1r[p] + i_ * g_;
            c1r[p] = c;
            const float h = o_ * tanh_fast(c);
            out[(size_t)s * kB * kH + b * kH + col] = h;       // kernel output
            xh1nxt[b * kK2 + kH + col] = f2bf(h);              // bayes A / next L1 h
        }
        grid_barrier(bar, gen);

        // -------- Bayesian linear: x_{t+1} = h1 @ (mu + sig*eps)^T + b_t --------
        {   // build 16 x 512 bf16 weight tile for this WG's output columns
            const int jj = tid >> 5;                 // 0..15
            const int kb = (tid & 31) * 16;          // 0..496
            const int j  = 16 * g + jj;
            const float* ep = eps_w + (size_t)s * kH * kH + (size_t)j * kH + kb;
            const float* mp = w_mu + (size_t)j * kH + kb;
            const float* sp = sigw + (size_t)j * kH + kb;
            unsigned short* wp = wtile + jj * kKP2 + kb;
#pragma unroll
            for (int e = 0; e < 16; e += 4) {
                // eps_w is streamed exactly once -> non-temporal (don't thrash L2)
                const v4f ev = __builtin_nontemporal_load((const v4f*)(ep + e));
                const v4f mv = *(const v4f*)(mp + e);
                const v4f sv = *(const v4f*)(sp + e);
                wp[e + 0] = f2bf(mv.x + sv.x * ev.x);
                wp[e + 1] = f2bf(mv.y + sv.y * ev.y);
                wp[e + 2] = f2bf(mv.z + sv.z * ev.z);
                wp[e + 3] = f2bf(mv.w + sv.w * ev.w);
            }
            if (tid < 16) {
                const int jb = 16 * g + tid;
                bb[tid] = b_mu[jb] + bsig[jb] *
                          __builtin_nontemporal_load(eps_b + (size_t)s * kH + jb);
            }
        }
        __syncthreads();
        if (wv < 4) {                                // 4 batch tiles, K = 512
            const int mt = wv;
            const int hi = lane >> 4;
            const unsigned short* A0 = xh1nxt + (16 * mt + (lane & 15)) * kK2 + kH + hi * 8;
            const unsigned short* B0 = wtile + (lane & 15) * kKP2 + hi * 8;
            v8f acc = {};
            v16bf aA = ldfrag(A0);
            v16bf bA = ldfrag(B0);
#pragma unroll 2
            for (int kk = 0; kk < kH; kk += 64) {
                v16bf aB = ldfrag(A0 + kk + 32);
                v16bf bB = ldfrag(B0 + kk + 32);
                acc = __builtin_amdgcn_wmma_f32_16x16x32_bf16(false, aA, false, bA,
                                                              (short)0, acc, false, false);
                if (kk + 64 < kH) {
                    aA = ldfrag(A0 + kk + 64);
                    bA = ldfrag(B0 + kk + 64);
                }
                acc = __builtin_amdgcn_wmma_f32_16x16x32_bf16(false, aB, false, bB,
                                                              (short)0, acc, false, false);
            }
            const int jj = lane & 15;
            const int mb = 16 * mt + hi * 8;
#pragma unroll
            for (int v = 0; v < 8; ++v)
                xh0nxt[(mb + v) * kK2 + 16 * g + jj] = f2bf(acc[v] + bb[jj]);
        }
        grid_barrier(bar, gen);
    }
}

extern "C" void kernel_launch(void* const* d_in, const int* in_sizes, int n_in,
                              void* d_out, int out_size, void* d_ws, size_t ws_size,
                              hipStream_t stream) {
    (void)in_sizes; (void)n_in; (void)out_size; (void)ws_size;
    const float* init_h    = (const float*)d_in[1];
    const float* init_c    = (const float*)d_in[2];
    const float* eps_w     = (const float*)d_in[3];
    const float* eps_b     = (const float*)d_in[4];
    const float* W_ih0     = (const float*)d_in[5];
    const float* W_hh0     = (const float*)d_in[6];
    const float* b_ih0     = (const float*)d_in[7];
    const float* b_hh0     = (const float*)d_in[8];
    const float* W_ih1     = (const float*)d_in[9];
    const float* W_hh1     = (const float*)d_in[10];
    const float* b_ih1     = (const float*)d_in[11];
    const float* b_hh1     = (const float*)d_in[12];
    const float* w_mu      = (const float*)d_in[13];
    const float* w_log_var = (const float*)d_in[14];
    const float* b_mu      = (const float*)d_in[15];
    const float* b_log_var = (const float*)d_in[16];

    char* ws = (char*)d_ws;
    unsigned*       bar  = (unsigned*)ws;                       // zeroed by prep
    unsigned short* xh0  = (unsigned short*)(ws + 256);         // 2 x 128 KB
    unsigned short* xh1  = (unsigned short*)(ws + 256 + 2 * kXHN * 2);
    float*          sigw = (float*)(ws + 256 + 4 * kXHN * 2);   // H*H f32
    float*          bsig = (float*)(ws + 256 + 4 * kXHN * 2 + kH * kH * 4);

    (void)hipFuncSetAttribute((const void*)lstm_persistent,
                              hipFuncAttributeMaxDynamicSharedMemorySize, kSMEM);

    prep_kernel<<<(kH * kH + 255) / 256, 256, 0, stream>>>(w_log_var, b_log_var,
                                                           sigw, bsig, bar);
    lstm_persistent<<<kNWG, kTPB, kSMEM, stream>>>(
        init_h, init_c, eps_w, eps_b,
        W_ih0, W_hh0, b_ih0, b_hh0,
        W_ih1, W_hh1, b_ih1, b_hh1,
        w_mu, b_mu, sigw, bsig,
        bar, xh0, xh1, (float*)d_out);
}
